// Generator_59227599012197
// MI455X (gfx1250) — matrix operand order repaired and checked by reading, not compiled, not run on hardware
//
#include <hip/hip_runtime.h>
#include <math.h>

typedef __attribute__((ext_vector_type(16))) _Float16 v16h;
typedef __attribute__((ext_vector_type(8)))  _Float16 v8h;
typedef __attribute__((ext_vector_type(8)))  float    v8f;

#define BB  1024   // batch
#define TT  256    // time steps
#define HH  128    // hidden
#define ZZ  64     // init noise
#define MHW 256    // mlp width
#define DD  8      // data dim

// f32 LDS row stride (floats)
#define S1  132    // H=128-wide f32 state arrays
// f16 LDS row strides (halves, multiples of 8 for 16B-aligned v8h loads)
#define SHH 264    // MH=256-wide f16 activations
#define SY  136    // H=128-wide f16 activations
#define SZ16 72    // Z0=64-wide f16 staging

// Bias region offsets (floats, inside LDS)
enum : int {
  BO_IB1 = 0,    BO_IB2 = 256,  BO_FB1 = 384,  BO_FB2 = 640,  BO_FB3 = 896,
  BO_GB1 = 1024, BO_GB2 = 1280, BO_GB3 = 1536, BO_RB1 = 1664, BO_RB2 = 1920,
  BO_FT  = 1936, BO_GT  = 2192, BIAS_FLOATS = 2448
};

// Workspace layout (halves). Repacked matrix: (K/32 x Npad/16) blocks of 512
// halves; block (bk,bn) at (bn*nbk+bk)*512; in-block: half = col*32 + k.
enum : unsigned {
  OFF_IW1 = 0,
  OFF_IW2 = OFF_IW1 + 64u * 256u,
  OFF_FW1 = OFF_IW2 + 256u * 128u,   // rows 1..128 of fW1
  OFF_FW2 = OFF_FW1 + 128u * 256u,
  OFF_FW3 = OFF_FW2 + 256u * 256u,
  OFF_GW1 = OFF_FW3 + 256u * 128u,
  OFF_GW2 = OFF_GW1 + 128u * 256u,
  OFF_GW3 = OFF_GW2 + 256u * 256u,
  OFF_RW1 = OFF_GW3 + 256u * 128u,
  OFF_RW2 = OFF_RW1 + 128u * 256u,   // rW2 padded N: 8 -> 16
  WS_HALVES = OFF_RW2 + 256u * 16u
};

// ---------------------------------------------------------------------------
// Branch-free tanh: sign(x) * (1 - 2/(exp(2|x|)+1)).
// exp overflow -> +inf -> rcp -> 0 -> result saturates to +/-1; no branches.
// ---------------------------------------------------------------------------
__device__ __forceinline__ float tanh_fast(float x) {
  float ax = fabsf(x);
  float e  = __expf(2.0f * ax);                       // native v_exp path
  float r  = 1.0f - 2.0f * __builtin_amdgcn_rcpf(e + 1.0f);
  return copysignf(r, x);
}

__device__ __forceinline__ v16h loadA16(const _Float16* __restrict__ p) {
  v8h lo = *(const v8h*)p;          // K k0..k0+7   (lo lanes) / +8 (hi lanes)
  v8h hh = *(const v8h*)(p + 16);   // K k0+16..+23 (lo lanes) / +8 (hi lanes)
  v16h a;
#pragma unroll
  for (int j = 0; j < 8; ++j) { a[j] = lo[j]; a[8 + j] = hh[j]; }
  return a;
}

__device__ __forceinline__ v16h loadB16(const _Float16* __restrict__ p) {
  v8h lo = *(const v8h*)p;
  v8h hh = *(const v8h*)(p + 16);
  v16h b;
#pragma unroll
  for (int j = 0; j < 8; ++j) { b[j] = lo[j]; b[8 + j] = hh[j]; }
  return b;
}

// ---------------------------------------------------------------------------
// Repack: f32 row-major [K x Nsrc] -> f16 WMMA-B-fragment blocks [K x Npad]
// ---------------------------------------------------------------------------
__global__ void sde_repack_kernel(const float* __restrict__ src, int K, int Nsrc,
                                  _Float16* __restrict__ dst, int total) {
  int tid = blockIdx.x * blockDim.x + threadIdx.x;
  if (tid >= total) return;
  int nbk   = K >> 5;
  int block = tid >> 9;
  int r     = tid & 511;
  int col   = r >> 5;
  int kk    = r & 31;
  int bn    = block / nbk;
  int bk    = block % nbk;
  int n     = bn * 16 + col;
  int k     = bk * 32 + kk;
  float v = (n < Nsrc) ? src[(size_t)k * Nsrc + n] : 0.0f;
  dst[tid] = (_Float16)v;
}

// ---------------------------------------------------------------------------
// One 16-row WMMA layer. Input: f16 LDS (stride Sin halves). Each wave holds
// one A-fragment per k-step and feeds 1 or 2 N-tile accumulators (Ntiles/8).
// Output: optional f32 LDS and/or f16 LDS copies. Bias from LDS; optional
// time-fold bias_eff[n] = bias[n] + t1*trow[n].
// ---------------------------------------------------------------------------
__device__ __forceinline__ void layer16(
    const _Float16* __restrict__ in, int Sin, int K,
    const _Float16* __restrict__ W,
    const float* __restrict__ bias,
    const float* __restrict__ trow, float t1,
    float* __restrict__ outF, int SoutF,
    _Float16* __restrict__ outH, int SoutH,
    int Ntiles, bool doTanh, int wave, int lane)
{
  const int col = lane & 15;
  const int hi  = lane >> 4;
  const int nbk = K >> 5;
  const int nt  = Ntiles >> 3;           // 1 or 2 tiles per wave
  const int bn0 = wave;
  const int bn1 = wave + 8;

  float bv0 = bias[bn0 * 16 + col];
  if (trow) bv0 += t1 * trow[bn0 * 16 + col];
  v8f c0, c1;
#pragma unroll
  for (int i = 0; i < 8; ++i) c0[i] = bv0;
  if (nt == 2) {
    float bv1 = bias[bn1 * 16 + col];
    if (trow) bv1 += t1 * trow[bn1 * 16 + col];
#pragma unroll
    for (int i = 0; i < 8; ++i) c1[i] = bv1;
  }

  const _Float16* abase = in + col * Sin + hi * 8;
#pragma unroll
  for (int bk = 0; bk < nbk; ++bk) {
    v16h a = loadA16(abase + bk * 32);
    v16h b0 = loadB16(W + ((unsigned)(bn0 * nbk + bk) << 9) + col * 32 + hi * 8);
    c0 = __builtin_amdgcn_wmma_f32_16x16x32_f16(false, a, false, b0, (short)0, c0,
                                                false, false);
    if (nt == 2) {
      v16h b1 = loadB16(W + ((unsigned)(bn1 * nbk + bk) << 9) + col * 32 + hi * 8);
      c1 = __builtin_amdgcn_wmma_f32_16x16x32_f16(false, a, false, b1, (short)0, c1,
                                                  false, false);
    }
  }

  const int rbase = hi ? 8 : 0;
  {
    const int n = bn0 * 16 + col;
#pragma unroll
    for (int r = 0; r < 8; ++r) {
      float v = c0[r];
      if (doTanh) v = tanh_fast(v);
      if (outF) outF[(r + rbase) * SoutF + n] = v;
      if (outH) outH[(r + rbase) * SoutH + n] = (_Float16)v;
    }
  }
  if (nt == 2) {
    const int n = bn1 * 16 + col;
#pragma unroll
    for (int r = 0; r < 8; ++r) {
      float v = c1[r];
      if (doTanh) v = tanh_fast(v);
      if (outF) outF[(r + rbase) * SoutF + n] = v;
      if (outH) outH[(r + rbase) * SoutH + n] = (_Float16)v;
    }
  }
}

// ---------------------------------------------------------------------------
// Persistent-per-tile SDE kernel: each block owns 16 batch rows and runs the
// full reversible-Heun scan + fused readout.
// ---------------------------------------------------------------------------
extern "C" __global__ void __launch_bounds__(256)
sde_scan_kernel(const float* __restrict__ init_noise, const float* __restrict__ ts,
                const float* __restrict__ bm,
                const float* __restrict__ ib1, const float* __restrict__ ib2,
                const float* __restrict__ fW1, const float* __restrict__ fb1,
                const float* __restrict__ fb2, const float* __restrict__ fb3,
                const float* __restrict__ gW1, const float* __restrict__ gb1,
                const float* __restrict__ gb2, const float* __restrict__ gb3,
                const float* __restrict__ rb1, const float* __restrict__ rb2,
                const _Float16* __restrict__ ws, float* __restrict__ out)
{
  extern __shared__ float smem[];
  float* y   = smem;              // [16][S1] f32 state y_t
  float* yh  = y   + 16 * S1;     // [16][S1] f32 \hat{y}_t
  float* fA  = yh  + 16 * S1;     // f ping
  float* fBf = fA  + 16 * S1;     // f pong
  float* gA  = fBf + 16 * S1;     // g ping
  float* gB  = gA  + 16 * S1;     // g pong
  float* biasLds = gB + 16 * S1;  // BIAS_FLOATS
  _Float16* hA16 = (_Float16*)(biasLds + BIAS_FLOATS);  // [16][SHH]
  _Float16* hB16 = hA16 + 16 * SHH;                     // [16][SHH] (also z16)
  _Float16* y16  = hB16 + 16 * SHH;                     // [16][SY]
  _Float16* yh16 = y16 + 16 * SY;                       // [16][SY]

  const int tid  = threadIdx.x;
  const int wave = tid >> 5;
  const int lane = tid & 31;
  const int b0   = blockIdx.x * 16;

  const float t0v = ts[0];
  const float dtv = ts[1] - ts[0];
  const float sq  = sqrtf(dtv);

  // ---- stage biases + time-rows into LDS, z into f16 ----
  auto cpb = [&](int off, const float* src, int n) {
    for (int i = tid; i < n; i += 256) biasLds[off + i] = src[i];
  };
  cpb(BO_IB1, ib1, 256);  cpb(BO_IB2, ib2, 128);
  cpb(BO_FB1, fb1, 256);  cpb(BO_FB2, fb2, 256);  cpb(BO_FB3, fb3, 128);
  cpb(BO_GB1, gb1, 256);  cpb(BO_GB2, gb2, 256);  cpb(BO_GB3, gb3, 128);
  cpb(BO_RB1, rb1, 256);
  cpb(BO_FT, fW1, 256);   // fW1 row 0 (t coefficient)
  cpb(BO_GT, gW1, 256);   // gW1 row 0
  if (tid < 16) biasLds[BO_RB2 + tid] = (tid < DD) ? rb2[tid] : 0.0f;
  for (int idx = tid; idx < 16 * ZZ; idx += 256) {
    int rr = idx >> 6, cc = idx & 63;
    hB16[rr * SZ16 + cc] = (_Float16)init_noise[(size_t)(b0 + rr) * ZZ + cc];
  }
  __syncthreads();

  // drift+diffusion MLPs at time t1 on f16 input xin -> f32 (fo, go)
  auto fg_block = [&](float t1, const _Float16* xin, float* fo, float* go) {
    layer16(xin, SY, HH,  ws + OFF_FW1, biasLds + BO_FB1, biasLds + BO_FT, t1,
            nullptr, 0, hA16, SHH, 16, true, wave, lane);
    __syncthreads();
    layer16(hA16, SHH, MHW, ws + OFF_FW2, biasLds + BO_FB2, nullptr, 0.f,
            nullptr, 0, hB16, SHH, 16, true, wave, lane);
    __syncthreads();
    layer16(hB16, SHH, MHW, ws + OFF_FW3, biasLds + BO_FB3, nullptr, 0.f,
            fo, S1, nullptr, 0, 8, false, wave, lane);
    __syncthreads();
    layer16(xin, SY, HH,  ws + OFF_GW1, biasLds + BO_GB1, biasLds + BO_GT, t1,
            nullptr, 0, hA16, SHH, 16, true, wave, lane);
    __syncthreads();
    layer16(hA16, SHH, MHW, ws + OFF_GW2, biasLds + BO_GB2, nullptr, 0.f,
            nullptr, 0, hB16, SHH, 16, true, wave, lane);
    __syncthreads();
    layer16(hB16, SHH, MHW, ws + OFF_GW3, biasLds + BO_GB3, nullptr, 0.f,
            go, S1, nullptr, 0, 8, false, wave, lane);
    __syncthreads();
  };

  // fused readout of current y16 -> out[b0..b0+15, t, :]
  auto readout = [&](int t) {
    layer16(y16, SY, HH, ws + OFF_RW1, biasLds + BO_RB1, nullptr, 0.f,
            nullptr, 0, hA16, SHH, 16, true, wave, lane);
    __syncthreads();
    if (wave == 0) {  // single 16x16 tile (D=8 padded), K=256
      const int col = lane & 15;
      const int hi  = lane >> 4;
      float bv = biasLds[BO_RB2 + col];
      v8f c;
#pragma unroll
      for (int i = 0; i < 8; ++i) c[i] = bv;
      const _Float16* abase = hA16 + col * SHH + hi * 8;
#pragma unroll
      for (int bk = 0; bk < 8; ++bk) {
        v16h a = loadA16(abase + bk * 32);
        v16h b = loadB16(ws + OFF_RW2 + ((unsigned)bk << 9) + col * 32 + hi * 8);
        c = __builtin_amdgcn_wmma_f32_16x16x32_f16(false, a, false, b, (short)0, c,
                                                   false, false);
      }
      if (col < DD) {
        const int rbase = hi ? 8 : 0;
#pragma unroll
        for (int r = 0; r < 8; ++r)
          out[((size_t)(b0 + r + rbase) * TT + t) * DD + col] = c[r];
      }
    }
    __syncthreads();
  };

  // ----- initial embedding: x0 = tanh(z @ iW1 + ib1) @ iW2 + ib2 -----
  layer16(hB16, SZ16, ZZ, ws + OFF_IW1, biasLds + BO_IB1, nullptr, 0.f,
          nullptr, 0, hA16, SHH, 16, true, wave, lane);
  __syncthreads();
  layer16(hA16, SHH, MHW, ws + OFF_IW2, biasLds + BO_IB2, nullptr, 0.f,
          y, S1, y16, SY, 8, false, wave, lane);
  __syncthreads();
#pragma unroll
  for (int i = 0; i < 8; ++i) {  // yh0 = x0
    int idx = tid + i * 256;
    int rr = idx >> 7, cc = idx & 127;
    yh[rr * S1 + cc] = y[rr * S1 + cc];
  }
  __syncthreads();

  float* fp = fA; float* fn = fBf;
  float* gp = gA; float* gn = gB;

  fg_block(t0v, y16, fp, gp);   // f0, g0 at (ts[0], x0)
  readout(0);

  // ----- reversible Heun scan -----
  for (int tt = 1; tt < TT; ++tt) {
    const float t1 = ts[tt];
    float dw[8];
    // phase 1: yh1 = 2*y0 - yh0 + fp*dt + gp*dW  (dW kept in registers)
#pragma unroll
    for (int i = 0; i < 8; ++i) {
      int idx = tid + i * 256;
      int rr = idx >> 7, cc = idx & 127;
      dw[i] = bm[((size_t)(tt - 1) * BB + (b0 + rr)) * HH + cc] * sq;
      int o = rr * S1 + cc;
      float v = 2.0f * y[o] - yh[o] + fp[o] * dtv + gp[o] * dw[i];
      yh[o] = v;
      yh16[rr * SY + cc] = (_Float16)v;
    }
    __syncthreads();

    fg_block(t1, yh16, fn, gn);  // f1, g1 at (t1, yh1)

    // phase 2: y1 = y0 + 0.5*dt*(fp+f1) + 0.5*(gp+g1)*dW
#pragma unroll
    for (int i = 0; i < 8; ++i) {
      int idx = tid + i * 256;
      int rr = idx >> 7, cc = idx & 127;
      int o = rr * S1 + cc;
      float v = y[o] + 0.5f * dtv * (fp[o] + fn[o]) + 0.5f * (gp[o] + gn[o]) * dw[i];
      y[o] = v;
      y16[rr * SY + cc] = (_Float16)v;
    }
    __syncthreads();

    readout(tt);

    float* tp;
    tp = fp; fp = fn; fn = tp;
    tp = gp; gp = gn; gn = tp;
  }
}

// ---------------------------------------------------------------------------
extern "C" void kernel_launch(void* const* d_in, const int* in_sizes, int n_in,
                              void* d_out, int out_size, void* d_ws, size_t ws_size,
                              hipStream_t stream)
{
  const float* init_noise = (const float*)d_in[0];
  const float* ts  = (const float*)d_in[1];
  const float* bm  = (const float*)d_in[2];
  const float* iW1 = (const float*)d_in[3];  const float* ib1 = (const float*)d_in[4];
  const float* iW2 = (const float*)d_in[5];  const float* ib2 = (const float*)d_in[6];
  const float* fW1 = (const float*)d_in[7];  const float* fb1 = (const float*)d_in[8];
  const float* fW2 = (const float*)d_in[9];  const float* fb2 = (const float*)d_in[10];
  const float* fW3 = (const float*)d_in[11]; const float* fb3 = (const float*)d_in[12];
  const float* gW1 = (const float*)d_in[13]; const float* gb1 = (const float*)d_in[14];
  const float* gW2 = (const float*)d_in[15]; const float* gb2 = (const float*)d_in[16];
  const float* gW3 = (const float*)d_in[17]; const float* gb3 = (const float*)d_in[18];
  const float* rW1 = (const float*)d_in[19]; const float* rb1 = (const float*)d_in[20];
  const float* rW2 = (const float*)d_in[21]; const float* rb2 = (const float*)d_in[22];

  if (ws_size < (size_t)WS_HALVES * sizeof(_Float16)) return;
  _Float16* ws = (_Float16*)d_ws;

  auto rp = [&](const float* src, int K, int Nsrc, int Npad, unsigned off) {
    int total = K * Npad;  // halves
    int th = 256, blk = (total + th - 1) / th;
    sde_repack_kernel<<<blk, th, 0, stream>>>(src, K, Nsrc, ws + off, total);
  };
  rp(iW1,       64,  256, 256, OFF_IW1);
  rp(iW2,       256, 128, 128, OFF_IW2);
  rp(fW1 + 256, 128, 256, 256, OFF_FW1);  // skip t-row (folded into bias)
  rp(fW2,       256, 256, 256, OFF_FW2);
  rp(fW3,       256, 128, 128, OFF_FW3);
  rp(gW1 + 256, 128, 256, 256, OFF_GW1);
  rp(gW2,       256, 256, 256, OFF_GW2);
  rp(gW3,       256, 128, 128, OFF_GW3);
  rp(rW1,       128, 256, 256, OFF_RW1);
  rp(rW2,       256, 8,   16,  OFF_RW2);

  size_t lds_bytes = (size_t)(6 * 16 * S1 + BIAS_FLOATS) * sizeof(float)
                   + (size_t)(2 * 16 * SHH + 2 * 16 * SY) * sizeof(_Float16); // ~84 KB
  sde_scan_kernel<<<BB / 16, 256, lds_bytes, stream>>>(
      init_noise, ts, bm, ib1, ib2,
      fW1, fb1, fb2, fb3,
      gW1, gb1, gb2, gb3,
      rb1, rb2, ws, (float*)d_out);
}
